// EncoderBiLSTM_74174085202235
// MI455X (gfx1250) — matrix operand order repaired and checked by reading, not compiled, not run on hardware
//
#include <hip/hip_runtime.h>
#include <hip/hip_bf16.h>
#include <math.h>

// Problem constants (from reference)
#define B_    64
#define T_    512
#define D_    1024
#define H_    512
#define G4H_  2048
#define L_    2

typedef __bf16 v16bf __attribute__((ext_vector_type(16)));
typedef float  v8f   __attribute__((ext_vector_type(8)));
typedef unsigned int u32x4 __attribute__((ext_vector_type(4)));

union BF16Frag { v16bf v; u32x4 q[2]; };

__device__ __forceinline__ unsigned short f2bf(float f) {
  union { float f; unsigned u; } x; x.f = f;
  unsigned r = x.u + 0x7FFFu + ((x.u >> 16) & 1u);   // round-to-nearest-even
  return (unsigned short)(r >> 16);
}
__device__ __forceinline__ float sigm(float x) { return 1.0f / (1.0f + __expf(-x)); }

// CDNA5 async global->LDS copy (ASYNCcnt tracked, no VGPR round trip).
// LDS byte offset from the vdst VGPR; global addr = 64-bit VGPR pair.
__device__ __forceinline__ void async_g2l_b128(unsigned lds_off, const void* gptr) {
  asm volatile("global_load_async_to_lds_b128 %0, %1, off"
               :: "v"(lds_off), "v"((unsigned long long)(size_t)gptr)
               : "memory");
}
__device__ __forceinline__ void wait_async0() {
  asm volatile("s_wait_asynccnt 0x0" ::: "memory");
}

// ---------------------------------------------------------------------------
// One-time prep: fp32 -> bf16 weights, fused bias, bf16 copy of initial h.
// ---------------------------------------------------------------------------
__global__ void prep_kernel(const float* __restrict__ Wih, const float* __restrict__ Whh,
                            const float* __restrict__ bih, const float* __restrict__ bhh,
                            const float* __restrict__ h0,
                            unsigned short* __restrict__ wih_bf,
                            unsigned short* __restrict__ whh_bf,
                            float* __restrict__ bias,
                            unsigned short* __restrict__ hbf /* [L][parity=2][2][B][H], fill parity 0 */) {
  const long NW1 = (long)L_ * 2 * G4H_ * D_;     // 8388608
  const long NW2 = (long)L_ * 2 * G4H_ * H_;     // 4194304
  const long NB  = (long)L_ * 2 * G4H_;          // 8192
  const long NH  = (long)L_ * 2 * B_ * H_;       // 131072
  const long total = NW1 + NW2 + NB + NH;
  for (long i = (long)blockIdx.x * blockDim.x + threadIdx.x; i < total;
       i += (long)gridDim.x * blockDim.x) {
    if (i < NW1) {
      wih_bf[i] = f2bf(Wih[i]);
    } else if (i < NW1 + NW2) {
      long j = i - NW1;
      whh_bf[j] = f2bf(Whh[j]);
    } else if (i < NW1 + NW2 + NB) {
      long j = i - NW1 - NW2;
      bias[j] = bih[j] + bhh[j];
    } else {
      long j = i - NW1 - NW2 - NB;        // over [L][2][B][H]
      long l = j >> 16;                   // / (2*B*H = 65536)
      long rem = j & 65535;
      hbf[l * (2 * 2 * B_ * H_) + rem] = f2bf(h0[j]);   // parity 0 slot
    }
  }
}

// ---------------------------------------------------------------------------
// Embedding gather+sum -> bf16, stored [T][B][D]
// ---------------------------------------------------------------------------
__global__ void embed_kernel(const int* __restrict__ rt, const int* __restrict__ re,
                             const int* __restrict__ rm,
                             const float* __restrict__ Ert, const float* __restrict__ Ere,
                             const float* __restrict__ Erm,
                             unsigned short* __restrict__ x0) {
  const int NV8 = T_ * B_ * (D_ / 8);          // vec8 units
  for (int v = blockIdx.x * blockDim.x + threadIdx.x; v < NV8;
       v += gridDim.x * blockDim.x) {
    int c8 = v & (D_ / 8 - 1);                 // column block (d = c8*8)
    int bt = v >> 7;
    int b  = bt & (B_ - 1);
    int t  = bt >> 6;
    long tok0 = rt[b * T_ + t];
    long tok1 = re[b * T_ + t];
    long tok2 = rm[b * T_ + t];
    const float* p0 = Ert + tok0 * D_ + c8 * 8;
    const float* p1 = Ere + tok1 * D_ + c8 * 8;
    const float* p2 = Erm + tok2 * D_ + c8 * 8;
    float4 a0 = *(const float4*)(p0), a1 = *(const float4*)(p0 + 4);
    float4 b0 = *(const float4*)(p1), b1 = *(const float4*)(p1 + 4);
    float4 d0 = *(const float4*)(p2), d1 = *(const float4*)(p2 + 4);
    float s[8];
    s[0] = a0.x + b0.x + d0.x; s[1] = a0.y + b0.y + d0.y;
    s[2] = a0.z + b0.z + d0.z; s[3] = a0.w + b0.w + d0.w;
    s[4] = a1.x + b1.x + d1.x; s[5] = a1.y + b1.y + d1.y;
    s[6] = a1.z + b1.z + d1.z; s[7] = a1.w + b1.w + d1.w;
    union { unsigned short h[8]; u32x4 q; } ov;
#pragma unroll
    for (int r = 0; r < 8; ++r) ov.h[r] = f2bf(s[r]);
    *(u32x4*)(x0 + (long)v * 8) = ov.q;
  }
}

// ---------------------------------------------------------------------------
// One LSTM step for one layer (both directions via blockIdx.y).
// Block tile: for its direction, gates for all 4 gate-chunks over 32 columns
// of H and all 64 batch rows. 8 waves: wave -> (gate = w>>1, joff = (w&1)*16).
// ---------------------------------------------------------------------------
#define XPITCH 1032                        // 1024 + 8 bf16 pad (bank spread)
#define HPITCH 520                         // 512 + 8
#define GPITCH 33
#define SMEM_X_BYTES (B_ * XPITCH * 2)                 // 132096
#define SMEM_H_OFF   SMEM_X_BYTES
#define SMEM_H_BYTES (B_ * HPITCH * 2)                 // 66560
#define SMEM_G_OFF   (SMEM_H_OFF + SMEM_H_BYTES)       // 198656
#define SMEM_G_BYTES (4 * B_ * GPITCH * 4)             // 33792
#define SMEM_TOTAL   (SMEM_G_OFF + SMEM_G_BYTES)       // 232448 < 320K WGP LDS
static_assert(SMEM_TOTAL <= 320 * 1024, "LDS budget");

__global__ __launch_bounds__(256) void lstm_step_kernel(
    const unsigned short* __restrict__ xin,  // [B][D] bf16 layer input
    const unsigned short* __restrict__ Wx,   // [2][4H][D] bf16 (layer slice)
    const unsigned short* __restrict__ Wh,   // [2][4H][H] bf16
    const float* __restrict__ bias,          // [2][4H]
    float* __restrict__ cst,                 // [2][B][H] f32 (layer slice)
    float* __restrict__ hst,                 // [2][B][H] f32
    const unsigned short* __restrict__ hbf_rd, // [2][B][H] bf16 (prev h)
    unsigned short* __restrict__ hbf_wr,       // [2][B][H] bf16 (new h)
    unsigned short* __restrict__ xout,       // [B][D] bf16 (layer0) or null
    float* __restrict__ fout,                // d_out + t*D (layer1) or null
    int reset) {
  extern __shared__ __align__(16) unsigned char smem[];
  unsigned short* xs = (unsigned short*)smem;
  unsigned short* hs = (unsigned short*)(smem + SMEM_H_OFF);
  float* gb = (float*)(smem + SMEM_G_OFF);

  const int tid  = threadIdx.x;
  const int lane = tid & 31;
  const int wave = tid >> 5;
  const int dir  = blockIdx.y;
  const int j0   = blockIdx.x * 32;
  const int gate = wave >> 1;
  const int joff = (wave & 1) << 4;
  const int n0   = gate * H_ + j0 + joff;    // base column in 4H gate dim
  const int ml   = lane & 15;
  const int kh   = lane >> 4;

  // ---- async-stage x tile [64 x 1024] bf16 into LDS (shared by 8 waves) ----
  for (int idx = tid; idx < B_ * (D_ / 8); idx += 256) {
    int r = idx >> 7, c = (idx & 127) << 3;
    unsigned loff = (unsigned)(size_t)(xs + r * XPITCH + c);
    async_g2l_b128(loff, xin + r * D_ + c);
  }
  if (!reset) {
    const unsigned short* hsrc = hbf_rd + dir * (B_ * H_);
    for (int idx = tid; idx < B_ * (H_ / 8); idx += 256) {
      int r = idx >> 6, c = (idx & 63) << 3;
      unsigned loff = (unsigned)(size_t)(hs + r * HPITCH + c);
      async_g2l_b128(loff, hsrc + r * H_ + c);
    }
  }
  wait_async0();
  __syncthreads();

  // ---- accumulators preloaded with fused bias (depends only on N) ----
  const float bv = bias[dir * G4H_ + n0 + ml];
  v8f acc[4];
#pragma unroll
  for (int mt = 0; mt < 4; ++mt)
#pragma unroll
    for (int r = 0; r < 8; ++r) acc[mt][r] = bv;

  // ---- g += x @ Wih^T : K = 1024 ----
  {
    const unsigned short* wrow = Wx + ((long)(dir * G4H_ + n0 + ml)) * D_;
    for (int k0 = 0; k0 < D_; k0 += 32) {
      BF16Frag bfr;
      const unsigned short* wp = wrow + k0 + (kh << 4);  // lane=N col, 16 contig K
      bfr.q[0] = *(const u32x4*)(wp);
      bfr.q[1] = *(const u32x4*)(wp + 8);
      if (k0 + 32 < D_) __builtin_prefetch(wrow + k0 + 32, 0, 3);
      // load all four A fragments first, then issue 4 back-to-back WMMAs
      BF16Frag afr[4];  // lanes 0-15: K{0..15}, lanes 16-31: K{16..31}
#pragma unroll
      for (int mt = 0; mt < 4; ++mt) {
        const unsigned short* ap = xs + (mt * 16 + ml) * XPITCH + k0 + (kh << 4);
        afr[mt].q[0] = *(const u32x4*)(ap);
        afr[mt].q[1] = *(const u32x4*)(ap + 8);
      }
#pragma unroll
      for (int mt = 0; mt < 4; ++mt)
        acc[mt] = __builtin_amdgcn_wmma_f32_16x16x32_bf16(
            false, afr[mt].v, false, bfr.v, (short)0, acc[mt], false, false);
    }
  }

  // ---- g += h @ Whh^T : K = 512 (skipped when state is reset to zero) ----
  if (!reset) {
    const unsigned short* wrow = Wh + ((long)(dir * G4H_ + n0 + ml)) * H_;
    for (int k0 = 0; k0 < H_; k0 += 32) {
      BF16Frag bfr;
      const unsigned short* wp = wrow + k0 + (kh << 4);
      bfr.q[0] = *(const u32x4*)(wp);
      bfr.q[1] = *(const u32x4*)(wp + 8);
      if (k0 + 32 < H_) __builtin_prefetch(wrow + k0 + 32, 0, 3);
      BF16Frag afr[4];
#pragma unroll
      for (int mt = 0; mt < 4; ++mt) {
        const unsigned short* ap = hs + (mt * 16 + ml) * HPITCH + k0 + (kh << 4);
        afr[mt].q[0] = *(const u32x4*)(ap);
        afr[mt].q[1] = *(const u32x4*)(ap + 8);
      }
#pragma unroll
      for (int mt = 0; mt < 4; ++mt)
        acc[mt] = __builtin_amdgcn_wmma_f32_16x16x32_bf16(
            false, afr[mt].v, false, bfr.v, (short)0, acc[mt], false, false);
    }
  }

  // ---- gate exchange: C layout VGPR r -> M = r + (lane>=16)*8, N = lane&15 ----
#pragma unroll
  for (int mt = 0; mt < 4; ++mt)
#pragma unroll
    for (int r = 0; r < 8; ++r) {
      int m = mt * 16 + (kh << 3) + r;
      gb[(gate * B_ + m) * GPITCH + joff + ml] = acc[mt][r];
    }
  __syncthreads();

  // ---- elementwise LSTM cell update (each block owns j in [j0, j0+32)) ----
  for (int e = tid; e < B_ * 32; e += 256) {
    int b  = e >> 5;
    int jl = e & 31;
    int j  = j0 + jl;
    float gi = gb[(0 * B_ + b) * GPITCH + jl];
    float gf = gb[(1 * B_ + b) * GPITCH + jl];
    float gg = gb[(2 * B_ + b) * GPITCH + jl];
    float go = gb[(3 * B_ + b) * GPITCH + jl];
    long ci = (long)dir * (B_ * H_) + b * H_ + j;
    float cold = reset ? 0.0f : cst[ci];
    float cn = sigm(gf) * cold + sigm(gi) * tanhf(gg);
    float hn = sigm(go) * tanhf(cn);
    cst[ci] = cn;
    hst[ci] = hn;
    hbf_wr[ci] = f2bf(hn);
    if (xout) xout[b * D_ + dir * H_ + j] = f2bf(hn);
    if (fout) fout[(long)b * (T_ * D_) + dir * H_ + j] = hn;
  }
}

// ---------------------------------------------------------------------------
// Host-side orchestration
// ---------------------------------------------------------------------------
extern "C" void kernel_launch(void* const* d_in, const int* in_sizes, int n_in,
                              void* d_out, int out_size, void* d_ws, size_t ws_size,
                              hipStream_t stream) {
  (void)in_sizes; (void)n_in; (void)out_size; (void)ws_size;
  const int*   rt  = (const int*)d_in[0];
  const int*   re  = (const int*)d_in[1];
  const int*   rm  = (const int*)d_in[2];
  const float* h0  = (const float*)d_in[3];
  const float* c0  = (const float*)d_in[4];
  const float* Ert = (const float*)d_in[5];
  const float* Ere = (const float*)d_in[6];
  const float* Erm = (const float*)d_in[7];
  const float* Wih = (const float*)d_in[8];
  const float* Whh = (const float*)d_in[9];
  const float* bih = (const float*)d_in[10];
  const float* bhh = (const float*)d_in[11];
  float* out = (float*)d_out;

  // workspace layout (bytes)
  char* ws = (char*)d_ws;
  const size_t O_WIH  = 0;                         // 16,777,216 B
  const size_t O_WHH  = O_WIH + 16777216;          //  8,388,608 B
  const size_t O_BIAS = O_WHH + 8388608;           //     32,768 B
  const size_t O_HST  = O_BIAS + 32768;            //    524,288 B  [L][2][B][H] f32
  const size_t O_CST  = O_HST + 524288;            //    524,288 B
  const size_t O_HBF  = O_CST + 524288;            //    524,288 B  [L][par][2][B][H] bf16
  const size_t O_X1   = O_HBF + 524288;            //    131,072 B  [B][D] bf16
  const size_t O_X0   = O_X1 + 131072;             // 67,108,864 B  [T][B][D] bf16

  unsigned short* wih_bf = (unsigned short*)(ws + O_WIH);
  unsigned short* whh_bf = (unsigned short*)(ws + O_WHH);
  float*          bias   = (float*)(ws + O_BIAS);
  float*          hst    = (float*)(ws + O_HST);
  float*          cst    = (float*)(ws + O_CST);
  unsigned short* hbf    = (unsigned short*)(ws + O_HBF);
  unsigned short* x1     = (unsigned short*)(ws + O_X1);
  unsigned short* x0     = (unsigned short*)(ws + O_X0);

  const size_t STATE_BYTES = (size_t)L_ * 2 * B_ * H_ * 4;   // 524288
  hipMemcpyAsync(hst, h0, STATE_BYTES, hipMemcpyDeviceToDevice, stream);
  hipMemcpyAsync(cst, c0, STATE_BYTES, hipMemcpyDeviceToDevice, stream);

  prep_kernel<<<2048, 256, 0, stream>>>(Wih, Whh, bih, bhh, h0,
                                        wih_bf, whh_bf, bias, hbf);
  embed_kernel<<<4096, 256, 0, stream>>>(rt, re, rm, Ert, Ere, Erm, x0);

  const long WXL = (long)2 * G4H_ * D_;    // per-layer Wih stride (elems)
  const long WHL = (long)2 * G4H_ * H_;    // per-layer Whh stride
  const long BL  = (long)2 * G4H_;         // per-layer bias stride
  const long SL  = (long)2 * B_ * H_;      // per-layer state stride (elems)
  const long HBL = (long)2 * 2 * B_ * H_;  // per-layer hbf stride (parity*dir*B*H)
  const long HBP = (long)2 * B_ * H_;      // parity stride

  dim3 grid(H_ / 32, 2, 1);                // 16 column tiles x 2 directions
  for (int t = 0; t < T_; ++t) {
    int reset = (t > 0 && (t % 32) == 0) ? 1 : 0;
    int pr = t & 1;            // read parity
    int pw = pr ^ 1;           // write parity

    // layer 0: input = embeddings, output h -> x1 (layer-1 input)
    lstm_step_kernel<<<grid, 256, SMEM_TOTAL, stream>>>(
        x0 + (size_t)t * B_ * D_,
        wih_bf, whh_bf, bias, cst, hst,
        hbf + pr * HBP, hbf + pw * HBP,
        x1, (float*)nullptr, reset);

    // layer 1: input = x1, output h -> d_out[:, t, :]
    lstm_step_kernel<<<grid, 256, SMEM_TOTAL, stream>>>(
        x1,
        wih_bf + WXL, whh_bf + WHL, bias + BL, cst + SL, hst + SL,
        hbf + HBL + pr * HBP, hbf + HBL + pw * HBP,
        (unsigned short*)nullptr, out + (size_t)t * D_, reset);
  }

  // final states: d_out tail = hT [L*2,B,H] then cT [L*2,B,H]
  const size_t NOUT = (size_t)B_ * T_ * D_;
  hipMemcpyAsync(out + NOUT, hst, STATE_BYTES, hipMemcpyDeviceToDevice, stream);
  hipMemcpyAsync(out + NOUT + (size_t)L_ * 2 * B_ * H_, cst, STATE_BYTES,
                 hipMemcpyDeviceToDevice, stream);
}
